// Attention__86509231276504
// MI455X (gfx1250) — matrix operand order repaired and checked by reading, not compiled, not run on hardware
//
#include <hip/hip_runtime.h>

typedef __bf16 bf16;
typedef __bf16 bf16x16 __attribute__((ext_vector_type(16)));
typedef float f32x8 __attribute__((ext_vector_type(8)));

#define DN 0.3535533905932738f   /* 64^-0.25 */
#define RATIO 0.0625f            /* 256^-0.5 */
#define EPSK 1e-4f
#define EPSA 1e-6f

__device__ inline f32x8 wmma_bf16(bf16x16 a, bf16x16 b, f32x8 c) {
  return __builtin_amdgcn_wmma_f32_16x16x32_bf16(false, a, false, b, (short)0, c,
                                                 false, false);
}

// A-fragment (16x32 bf16): lane row fixed; lanes<16 hold K 0..7,16..23; lanes>=16 hold
// K 8..15,24..31. Caller passes p = rowbase + kk + (hi?8:0).
__device__ inline bf16x16 ldA(const bf16* p) {
  bf16x16 r;
#pragma unroll
  for (int i = 0; i < 8; ++i) { r[i] = p[i]; r[i + 8] = p[i + 16]; }
  return r;
}
// B-fragment (32x16 bf16): lane col fixed; lanes<16 hold K 0..7,16..23; lanes>=16 hold
// K 8..15,24..31. Caller passes p = colbase + kk + (hi?8:0).
__device__ inline bf16x16 ldB(const bf16* p) {
  bf16x16 r;
#pragma unroll
  for (int i = 0; i < 8; ++i) { r[i] = p[i]; r[i + 8] = p[i + 16]; }
  return r;
}

__device__ inline void zero8(f32x8& v) {
#pragma unroll
  for (int i = 0; i < 8; ++i) v[i] = 0.f;
}

__device__ inline void atomicMaxFloat(float* addr, float val) {
  unsigned* ua = (unsigned*)addr;
  unsigned old = *ua;
  while (true) {
    float f = __uint_as_float(old);
    if (f >= val) break;
    unsigned assumed = old;
    old = atomicCAS(ua, assumed, __float_as_uint(val));
    if (old == assumed) break;
  }
}

// ---------------- elementwise helpers ----------------
__global__ void cvt_kernel(const float* __restrict__ s, bf16* __restrict__ d, int n) {
  int i = blockIdx.x * 256 + threadIdx.x;
  if (i < n) d[i] = (bf16)s[i];
}

__global__ void init_kmax(float* kmax) { if (threadIdx.x == 0) kmax[0] = -3.0e38f; }

// diag[r] = 0.5 * sum(stored^2)  (stored already scaled by DN)
__global__ void diag_kernel(const bf16* __restrict__ qh, float* __restrict__ dg, int rows) {
  int r = blockIdx.x * 256 + threadIdx.x;
  if (r >= rows) return;
  const bf16* p = qh + (long)r * 64;
  float s = 0.f;
#pragma unroll 8
  for (int e = 0; e < 64; ++e) { float v = (float)p[e]; s += v * v; }
  dg[r] = 0.5f * s;
}

// ---------------- QKV projection: Y(heads) = alpha * X @ W^T ----------------
__global__ __launch_bounds__(256) void qkv_gemm(const bf16* __restrict__ X,
                                                const bf16* __restrict__ W,
                                                bf16* __restrict__ Y, float alpha) {
  int gw = (blockIdx.x * 256 + threadIdx.x) >> 5;
  int lane = threadIdx.x & 31;
  int jl = lane & 15, hi = lane >> 4;
  int nt = gw & 31, mt = gw >> 5;  // N=512 -> 32 col tiles
  int m0 = mt * 16, n0 = nt * 16;
  const bf16* arow = X + (long)(m0 + jl) * 512 + (hi ? 8 : 0);
  const bf16* brow = W + (long)(n0 + jl) * 512 + (hi ? 8 : 0);
  f32x8 acc; zero8(acc);
  for (int kk = 0; kk < 512; kk += 32)
    acc = wmma_bf16(ldA(arow + kk), ldB(brow + kk), acc);
#pragma unroll
  for (int c = 0; c < 8; ++c) {
    int m = m0 + c + 8 * hi;            // row in [8192)
    int n = n0 + jl;                    // col in [512)
    int bi = m >> 12, ni = m & 4095;
    int h = n >> 6, e = n & 63;
    Y[((long)(bi * 8 + h) * 4096 + ni) * 64 + e] = (bf16)(alpha * acc[c]);
  }
}

// ---------------- feature map: dd = data @ proj^T ; exp(dd - diag - m) ----------------
// mode 0: q (row max, store); mode 1: k pass1 (global max); mode 2: k pass2 (store)
__global__ __launch_bounds__(256) void feat_kernel(const bf16* __restrict__ Qh,
                                                   const bf16* __restrict__ projb,
                                                   const float* __restrict__ dg,
                                                   float* __restrict__ kmax,
                                                   bf16* __restrict__ out, int mode) {
  int gw = (blockIdx.x * 256 + threadIdx.x) >> 5;
  int lane = threadIdx.x & 31;
  int jl = lane & 15, hi = lane >> 4;
  int r0 = gw * 16;
  const bf16* arow = Qh + (long)(r0 + jl) * 64 + (hi ? 8 : 0);
  int boff = hi ? 8 : 0;
  f32x8 acc[16];
#pragma unroll
  for (int t = 0; t < 16; ++t) zero8(acc[t]);
#pragma unroll
  for (int kk = 0; kk < 64; kk += 32) {
    bf16x16 a = ldA(arow + kk);
#pragma unroll
    for (int t = 0; t < 16; ++t) {
      bf16x16 b = ldB(projb + (long)(t * 16 + jl) * 64 + kk + boff);
      acc[t] = wmma_bf16(a, b, acc[t]);
    }
  }
  if (mode == 1) {  // global max over everything
    float m = -3.0e38f;
#pragma unroll
    for (int t = 0; t < 16; ++t)
#pragma unroll
      for (int e = 0; e < 8; ++e) m = fmaxf(m, acc[t][e]);
    m = fmaxf(m, __shfl_xor(m, 16, 32));
    m = fmaxf(m, __shfl_xor(m, 8, 32));
    m = fmaxf(m, __shfl_xor(m, 4, 32));
    m = fmaxf(m, __shfl_xor(m, 2, 32));
    m = fmaxf(m, __shfl_xor(m, 1, 32));
    if (lane == 0) atomicMaxFloat(kmax, m);
    return;
  }
  float mrow[8];
  if (mode == 0) {
#pragma unroll
    for (int e = 0; e < 8; ++e) {
      float m = acc[0][e];
#pragma unroll
      for (int t = 1; t < 16; ++t) m = fmaxf(m, acc[t][e]);
      m = fmaxf(m, __shfl_xor(m, 8, 32));  // stay within 16-lane half
      m = fmaxf(m, __shfl_xor(m, 4, 32));
      m = fmaxf(m, __shfl_xor(m, 2, 32));
      m = fmaxf(m, __shfl_xor(m, 1, 32));
      mrow[e] = m;
    }
  } else {
    float gm = kmax[0];
#pragma unroll
    for (int e = 0; e < 8; ++e) mrow[e] = gm;
  }
#pragma unroll
  for (int e = 0; e < 8; ++e) {
    int row = r0 + e + 8 * hi;
    float d = dg[row];
#pragma unroll
    for (int t = 0; t < 16; ++t) {
      float o = RATIO * (__expf(acc[t][e] - d - mrow[e]) + EPSK);
      out[(long)row * 256 + t * 16 + jl] = (bf16)o;
    }
  }
}

// ---------------- causal linear-attention scan, one block per (b,h) ----------------
__global__ __launch_bounds__(256) void scan_kernel(const bf16* __restrict__ qf,
                                                   const bf16* __restrict__ kf,
                                                   const bf16* __restrict__ vh,
                                                   bf16* __restrict__ attnb,
                                                   float* __restrict__ dout) {
  __shared__ bf16 sKt[256 * 128];  // Kt[f][j]   64KB
  __shared__ bf16 sVt[64 * 128];   // Vt[e][j]   16KB
  __shared__ bf16 sA[128 * 128];   // masked QK^T 32KB
  __shared__ bf16 sSt[64 * 256];   // S^T bf16   32KB
  __shared__ float sZ[256];
  __shared__ float sD[128];

  const int tid = threadIdx.x;
  const int w = tid >> 5;
  const int lane = tid & 31;
  const int jl = lane & 15;
  const int hi = lane >> 4;
  const int bh = blockIdx.x;  // 0..15
  const int r0 = w * 16;
  const int aoff = hi ? 8 : 0;
  const int boff = hi ? 8 : 0;

  // persistent S accumulators: f-tiles {2w, 2w+1} x e-tiles {0..3}
  f32x8 sacc[2][4];
#pragma unroll
  for (int a = 0; a < 2; ++a)
#pragma unroll
    for (int t = 0; t < 4; ++t) zero8(sacc[a][t]);

  if (tid < 256) sZ[tid] = 0.f;
  for (int i = tid; i < 64 * 256; i += 256) sSt[i] = (bf16)0.f;
  __syncthreads();

  for (int ic = 0; ic < 32; ++ic) {
    const long base = (long)bh * 4096 + ic * 128;

    // --- stage K^T and V^T into LDS ---
    for (int x = tid; x < 128 * 256; x += 256) {
      int j = x >> 8, f = x & 255;
      sKt[f * 128 + j] = kf[(base + j) * 256 + f];
    }
    for (int x = tid; x < 128 * 64; x += 256) {
      int j = x >> 6, e = x & 63;
      sVt[e * 128 + j] = vh[(base + j) * 64 + e];
    }

    // --- A = Q K^T (global operands), mask + rowsum ---
    {
      f32x8 acc[8];
#pragma unroll
      for (int t = 0; t < 8; ++t) zero8(acc[t]);
      const bf16* arow = qf + (base + r0 + jl) * 256 + aoff;
      for (int kk = 0; kk < 256; kk += 32) {
        bf16x16 a = ldA(arow + kk);
#pragma unroll
        for (int t = 0; t < 8; ++t) {
          bf16x16 b = ldB(kf + (base + t * 16 + jl) * 256 + kk + boff);
          acc[t] = wmma_bf16(a, b, acc[t]);
        }
      }
      float rs[8];
#pragma unroll
      for (int e = 0; e < 8; ++e) rs[e] = 0.f;
#pragma unroll
      for (int t = 0; t < 8; ++t) {
#pragma unroll
        for (int e = 0; e < 8; ++e) {
          int il = r0 + e + 8 * hi;
          int jj = t * 16 + jl;
          float v = (jj <= il) ? acc[t][e] : 0.f;  // inclusive causal mask
          sA[il * 128 + jj] = (bf16)v;
          rs[e] += v;
        }
      }
#pragma unroll
      for (int e = 0; e < 8; ++e) {
        rs[e] += __shfl_xor(rs[e], 1, 32);
        rs[e] += __shfl_xor(rs[e], 2, 32);
        rs[e] += __shfl_xor(rs[e], 4, 32);
        rs[e] += __shfl_xor(rs[e], 8, 32);
      }
      if (jl == 0) {
#pragma unroll
        for (int e = 0; e < 8; ++e) sD[r0 + e + 8 * hi] = rs[e];
      }
    }
    __syncthreads();

    // --- D[i] += q_i . (Z + eps) ---
    if (tid < 128) {
      const bf16* qrow = qf + (base + tid) * 256;
      float s = 0.f;
      for (int f = 0; f < 256; ++f) s += (float)qrow[f] * (sZ[f] + EPSA);
      sD[tid] += s;
    }
    __syncthreads();

    // --- out = (Amask @ V + Q @ S_prev) / D ---
    {
      f32x8 acc[4];
#pragma unroll
      for (int t = 0; t < 4; ++t) zero8(acc[t]);
      const bf16* arow1 = sA + (r0 + jl) * 128 + aoff;
      for (int kk = 0; kk < 128; kk += 32) {
        bf16x16 a = ldA(arow1 + kk);
#pragma unroll
        for (int t = 0; t < 4; ++t) {
          bf16x16 b = ldB(sVt + (t * 16 + jl) * 128 + kk + boff);
          acc[t] = wmma_bf16(a, b, acc[t]);
        }
      }
      const bf16* arow2 = qf + (base + r0 + jl) * 256 + aoff;
      for (int kk = 0; kk < 256; kk += 32) {
        bf16x16 a = ldA(arow2 + kk);
#pragma unroll
        for (int t = 0; t < 4; ++t) {
          bf16x16 b = ldB(sSt + (t * 16 + jl) * 256 + kk + boff);
          acc[t] = wmma_bf16(a, b, acc[t]);
        }
      }
      const int bi = bh >> 3, h = bh & 7;
#pragma unroll
      for (int t = 0; t < 4; ++t) {
#pragma unroll
        for (int e = 0; e < 8; ++e) {
          int il = r0 + e + 8 * hi;
          float ov = acc[t][e] / sD[il];
          long pos = (long)bi * 4096 + ic * 128 + il;
          attnb[pos * 512 + h * 64 + t * 16 + jl] = (bf16)ov;
        }
      }
    }
    __syncthreads();

    // --- Z += colsum(K); S += K^T V (register accumulators); regen S^T bf16 ---
    if (tid < 256) {
      float z = sZ[tid];
      const bf16* kt = sKt + tid * 128;
#pragma unroll 8
      for (int j = 0; j < 128; ++j) z += (float)kt[j];
      sZ[tid] = z;
    }
#pragma unroll
    for (int fi = 0; fi < 2; ++fi) {
      const bf16* arow = sKt + ((2 * w + fi) * 16 + jl) * 128 + aoff;
      for (int kk = 0; kk < 128; kk += 32) {
        bf16x16 a = ldA(arow + kk);
#pragma unroll
        for (int t = 0; t < 4; ++t) {
          bf16x16 b = ldB(sVt + (t * 16 + jl) * 128 + kk + boff);
          sacc[fi][t] = wmma_bf16(a, b, sacc[fi][t]);
        }
      }
    }
#pragma unroll
    for (int fi = 0; fi < 2; ++fi)
#pragma unroll
      for (int t = 0; t < 4; ++t)
#pragma unroll
        for (int e = 0; e < 8; ++e) {
          int f = (2 * w + fi) * 16 + e + 8 * hi;
          int ee = t * 16 + jl;
          sSt[ee * 256 + f] = (bf16)sacc[fi][t][e];
        }
    __syncthreads();
  }

  // --- final states Zf [2,8,256], Sf [2,8,256,64] appended after out ---
  const long zoff = (long)2 * 4096 * 512;
  if (tid < 256) dout[zoff + bh * 256 + tid] = sZ[tid];
  const long soff = zoff + 4096;
#pragma unroll
  for (int fi = 0; fi < 2; ++fi)
#pragma unroll
    for (int t = 0; t < 4; ++t)
#pragma unroll
      for (int e = 0; e < 8; ++e) {
        int f = (2 * w + fi) * 16 + e + 8 * hi;
        int ee = t * 16 + jl;
        dout[soff + ((long)bh * 256 + f) * 64 + ee] = sacc[fi][t][e];
      }
}

// ---------------- final projection: out = attn @ Wo^T + bo (f32 out) ----------------
__global__ __launch_bounds__(256) void out_gemm(const bf16* __restrict__ A,
                                                const bf16* __restrict__ W,
                                                const float* __restrict__ bo,
                                                float* __restrict__ out) {
  int gw = (blockIdx.x * 256 + threadIdx.x) >> 5;
  int lane = threadIdx.x & 31;
  int jl = lane & 15, hi = lane >> 4;
  int nt = gw & 31, mt = gw >> 5;
  int m0 = mt * 16, n0 = nt * 16;
  const bf16* arow = A + (long)(m0 + jl) * 512 + (hi ? 8 : 0);
  const bf16* brow = W + (long)(n0 + jl) * 512 + (hi ? 8 : 0);
  f32x8 acc; zero8(acc);
  for (int kk = 0; kk < 512; kk += 32)
    acc = wmma_bf16(ldA(arow + kk), ldB(brow + kk), acc);
#pragma unroll
  for (int c = 0; c < 8; ++c) {
    int m = m0 + c + 8 * hi;
    int n = n0 + jl;
    out[(long)m * 512 + n] = acc[c] + bo[n];
  }
}

extern "C" void kernel_launch(void* const* d_in, const int* in_sizes, int n_in,
                              void* d_out, int out_size, void* d_ws, size_t ws_size,
                              hipStream_t stream) {
  const float* x = (const float*)d_in[0];
  const float* Wq = (const float*)d_in[1];
  const float* Wk = (const float*)d_in[2];
  const float* Wv = (const float*)d_in[3];
  const float* Wo = (const float*)d_in[4];
  const float* bo = (const float*)d_in[5];
  const float* proj = (const float*)d_in[6];
  float* out = (float*)d_out;

  const long M = 8192;      // b*n
  const long ROWS = 65536;  // b*h*n

  char* ws = (char*)d_ws;
  size_t off = 0;
  auto take = [&](size_t bytes) -> char* {
    char* p = ws + off;
    off += (bytes + 255) & ~(size_t)255;
    return p;
  };
  bf16* xb = (bf16*)take(M * 512 * 2);
  bf16* wqb = (bf16*)take(512 * 512 * 2);
  bf16* wkb = (bf16*)take(512 * 512 * 2);
  bf16* wvb = (bf16*)take(512 * 512 * 2);
  bf16* wob = (bf16*)take(512 * 512 * 2);
  bf16* projb = (bf16*)take(256 * 64 * 2);
  bf16* qh = (bf16*)take(ROWS * 64 * 2);
  bf16* kh = (bf16*)take(ROWS * 64 * 2);
  bf16* vh = (bf16*)take(ROWS * 64 * 2);
  float* diagq = (float*)take(ROWS * 4);
  float* diagk = (float*)take(ROWS * 4);
  float* kmax = (float*)take(256);
  bf16* qfeat = (bf16*)take(ROWS * 256 * 2);
  bf16* kfeat = (bf16*)take(ROWS * 256 * 2);
  bf16* attnb = (bf16*)take(M * 512 * 2);

  init_kmax<<<1, 64, 0, stream>>>(kmax);
  cvt_kernel<<<(int)((M * 512 + 255) / 256), 256, 0, stream>>>(x, xb, (int)(M * 512));
  cvt_kernel<<<1024, 256, 0, stream>>>(Wq, wqb, 512 * 512);
  cvt_kernel<<<1024, 256, 0, stream>>>(Wk, wkb, 512 * 512);
  cvt_kernel<<<1024, 256, 0, stream>>>(Wv, wvb, 512 * 512);
  cvt_kernel<<<1024, 256, 0, stream>>>(Wo, wob, 512 * 512);
  cvt_kernel<<<64, 256, 0, stream>>>(proj, projb, 256 * 64);

  // QKV projections (q,k pre-scaled by DN so diag/dd fold the normalizer in)
  qkv_gemm<<<2048, 256, 0, stream>>>(xb, wqb, qh, DN);
  qkv_gemm<<<2048, 256, 0, stream>>>(xb, wkb, kh, DN);
  qkv_gemm<<<2048, 256, 0, stream>>>(xb, wvb, vh, 1.0f);

  diag_kernel<<<256, 256, 0, stream>>>(qh, diagq, (int)ROWS);
  diag_kernel<<<256, 256, 0, stream>>>(kh, diagk, (int)ROWS);

  feat_kernel<<<512, 256, 0, stream>>>(qh, projb, diagq, kmax, qfeat, 0);  // q: rowmax
  feat_kernel<<<512, 256, 0, stream>>>(kh, projb, diagk, kmax, kfeat, 1);  // k: global max
  feat_kernel<<<512, 256, 0, stream>>>(kh, projb, diagk, kmax, kfeat, 2);  // k: apply

  scan_kernel<<<16, 256, 0, stream>>>(qfeat, kfeat, vh, attnb, out);

  out_gemm<<<2048, 256, 0, stream>>>(attnb, wob, bo, out);
}
